// BranchModule_15676630630683
// MI455X (gfx1250) — matrix operand order repaired and checked by reading, not compiled, not run on hardware
//
#include <hip/hip_runtime.h>
#include <hip/hip_bf16.h>
#include <math.h>

// ---------------------------------------------------------------------------
// Mamba 2-block stack for MI455X (gfx1250, wave32).
// GEMMs: v_wmma_f32_16x16x32_bf16, 2x4 tiles per wave, A staged in LDS via
// gfx1250 async-to-LDS (global_load_async_to_lds_b128 + s_wait_asynccnt).
// B loads / C stores forced to address space 1 -> global_load/store (pure
// LOADcnt/STOREcnt, keeps the LDS pipe free for ds_load + async DMA).
// ---------------------------------------------------------------------------

typedef __bf16 bf16_t;
typedef bf16_t v16bf __attribute__((ext_vector_type(16)));
typedef float  v8f   __attribute__((ext_vector_type(8)));

typedef const __attribute__((address_space(1))) bf16_t* gcb_p;   // global const bf16
typedef const __attribute__((address_space(1))) v16bf*  gcv_p;   // global const v16bf
typedef       __attribute__((address_space(1))) float*  gf_p;    // global float

#define DM    512    // D_MODEL
#define NBAT  2      // BATCH
#define NST   16     // D_STATE
#define DTR   32     // DT_RANK

__device__ __forceinline__ float silu_f(float v) { return v / (1.f + expf(-v)); }
__device__ __forceinline__ float softplus_f(float v) {
  return (v > 20.f) ? v : log1pf(expf(v));
}

// Low 32 bits of a generic pointer to __shared__ = LDS byte offset (ISA 10.2).
__device__ __forceinline__ unsigned lds_off_u32(const void* p) {
  return (unsigned)(unsigned long long)p;
}

// gfx1250 async DMA: one b128 per lane, global -> LDS, tracked by ASYNCcnt.
__device__ __forceinline__ void async_copy_b128(unsigned lds_byte_addr, const void* gaddr) {
  asm volatile("global_load_async_to_lds_b128 %0, %1, off"
               :: "v"(lds_byte_addr), "v"(gaddr) : "memory");
}
__device__ __forceinline__ void wait_async0() {
  asm volatile("s_wait_asynccnt 0" ::: "memory");
}

// ---------------------------------------------------------------------------
// LayerNorm over D for each (b,l): reads x in (B,D,L), writes (B*L, D) rows.
// ---------------------------------------------------------------------------
__global__ void ln_kernel(const float* __restrict__ x, const float* __restrict__ g,
                          const float* __restrict__ bb, float* __restrict__ out, int L) {
  int row = blockIdx.x * blockDim.x + threadIdx.x;   // row = b*L + l
  if (row >= NBAT * L) return;
  int b = row / L, l = row % L;
  const float* xp = x + (size_t)b * DM * L + l;
  float s = 0.f, s2 = 0.f;
  for (int d = 0; d < DM; ++d) {
    float v = xp[(size_t)d * L];
    s += v; s2 += v * v;
  }
  float mu  = s * (1.f / DM);
  float var = s2 * (1.f / DM) - mu * mu;
  float inv = rsqrtf(var + 1e-5f);
  float* op = out + (size_t)row * DM;
  for (int d = 0; d < DM; ++d)
    op[d] = (xp[(size_t)d * L] - mu) * inv * g[d] + bb[d];
}

// ---------------------------------------------------------------------------
// Pack f32 (M x K) activations (row stride ld) into WMMA A-fragment layout.
// ---------------------------------------------------------------------------
__global__ void pack_a_kernel(const float* __restrict__ src, bf16_t* __restrict__ dst,
                              int M, int K, int ld) {
  int idx = blockIdx.x * blockDim.x + threadIdx.x;
  if (idx >= M * K) return;
  int m = idx / K, k = idx % K;
  int mt = m >> 4, kt = k >> 5, kk = k & 31;
  int lane = (m & 15) + (((kk >> 3) & 1) << 4);
  int j    = ((kk & 7) >> 1) + ((kk & 16) ? 4 : 0);
  int half = kk & 1;
  int KT = K >> 5;
  size_t off = ((size_t)(mt * KT + kt)) * 512 + (size_t)lane * 16 + j * 2 + half;
  dst[off] = (bf16_t)src[(size_t)m * ld + k];
}

// ---------------------------------------------------------------------------
// Pack f32 (N x K) row-major weight (used as B = W^T) into WMMA B layout.
// ---------------------------------------------------------------------------
__global__ void pack_b_kernel(const float* __restrict__ w, bf16_t* __restrict__ dst,
                              int N, int K) {
  int idx = blockIdx.x * blockDim.x + threadIdx.x;
  if (idx >= N * K) return;
  int n = idx / K, k = idx % K;
  int nt = n >> 4, kt = k >> 5, kk = k & 31;
  int lane = (n & 15) + (((kk >> 3) & 1) << 4);
  int j    = ((kk & 7) >> 1) + ((kk & 16) ? 4 : 0);
  int half = kk & 1;
  int KT = K >> 5;
  size_t off = ((size_t)(nt * KT + kt)) * 512 + (size_t)lane * 16 + j * 2 + half;
  dst[off] = (bf16_t)w[(size_t)n * K + k];
}

// ---------------------------------------------------------------------------
// C(M,N) f32 = A(M,K)bf16 x B(K,N)bf16, pre-packed fragments.
// Block: 128 threads (4 waves), tile 32x256. Each wave: 2 M-tiles x 4 N-tiles
// = 8 WMMAs per k-step. A slab (2 KB/k-step) double-buffered in LDS via
// async-to-LDS DMA; B fragments streamed with global_load_b128 + prefetch.
// Software-pipelined main loop (no inner-loop branches) + peeled epilogue.
// ---------------------------------------------------------------------------
__global__ void __launch_bounds__(128)
wmma_gemm_bf16(const bf16_t* __restrict__ Ap, const bf16_t* __restrict__ Bp,
               float* __restrict__ C, int M, int N, int K) {
  __shared__ bf16_t sA[2][2 * 512];          // 2 buffers x (2 mtiles x 1KB)
  const int tid  = threadIdx.x;
  const int lane = tid & 31;
  const int wave = tid >> 5;
  const int KT = K >> 5;
  const int NT = N >> 4;
  const int mt0 = blockIdx.x * 2;
  const int ntBase = (blockIdx.y * 4 + wave) * 4;
  const bool valid = ntBase < NT;            // N is a multiple of 64 -> all-or-none

  // A staging: 128 threads x 16B = 2KB (one k-step, both mtiles)
  const int whichM = tid >> 6;               // 0/1 -> mtile
  const int off16  = tid & 63;               // 16B unit within 1KB tile
  const char* asrc = (const char*)Ap +
                     ((size_t)(mt0 + whichM) * KT) * 1024 + (size_t)off16 * 16;
  unsigned ldsA[2];
  ldsA[0] = lds_off_u32(&sA[0][whichM * 512]) + off16 * 16;
  ldsA[1] = lds_off_u32(&sA[1][whichM * 512]) + off16 * 16;

  async_copy_b128(ldsA[0], asrc);            // prologue: k-step 0

  // B rows in explicit global address space (global_load_b128, LOADcnt only)
  gcb_p brow[4];
#pragma unroll
  for (int j = 0; j < 4; ++j) {
    int nt = valid ? (ntBase + j) : 0;
    brow[j] = (gcb_p)(unsigned long long)Bp + ((size_t)nt * KT) * 512 + (size_t)lane * 16;
  }

  v8f acc[2][4];
#pragma unroll
  for (int m2 = 0; m2 < 2; ++m2)
#pragma unroll
    for (int j = 0; j < 4; ++j)
      acc[m2][j] = (v8f){0.f, 0.f, 0.f, 0.f, 0.f, 0.f, 0.f, 0.f};

  wait_async0();
  __syncthreads();

  for (int kt = 0; kt < KT - 1; ++kt) {      // pipelined steps (skipped if KT==1)
    async_copy_b128(ldsA[(kt + 1) & 1], asrc + (size_t)(kt + 1) * 1024);

    const bf16_t* ab = &sA[kt & 1][0];
    v16bf a0 = *reinterpret_cast<const v16bf*>(ab + lane * 16);          // ds_load_b128 x2
    v16bf a1 = *reinterpret_cast<const v16bf*>(ab + 512 + lane * 16);

#pragma unroll
    for (int j = 0; j < 4; ++j) {
      v16bf b = *(gcv_p)brow[j];
      __builtin_prefetch((const void*)(unsigned long long)(brow[j] + 512), 0, 3);
      acc[0][j] = __builtin_amdgcn_wmma_f32_16x16x32_bf16(false, a0, false, b,
                                                          (short)0, acc[0][j], false, false);
      acc[1][j] = __builtin_amdgcn_wmma_f32_16x16x32_bf16(false, a1, false, b,
                                                          (short)0, acc[1][j], false, false);
      brow[j] += 512;
    }
    wait_async0();                           // next A slab resident in LDS
    __syncthreads();
  }

  {                                          // epilogue: kt = KT-1
    const bf16_t* ab = &sA[(KT - 1) & 1][0];
    v16bf a0 = *reinterpret_cast<const v16bf*>(ab + lane * 16);
    v16bf a1 = *reinterpret_cast<const v16bf*>(ab + 512 + lane * 16);
#pragma unroll
    for (int j = 0; j < 4; ++j) {
      v16bf b = *(gcv_p)brow[j];
      acc[0][j] = __builtin_amdgcn_wmma_f32_16x16x32_bf16(false, a0, false, b,
                                                          (short)0, acc[0][j], false, false);
      acc[1][j] = __builtin_amdgcn_wmma_f32_16x16x32_bf16(false, a1, false, b,
                                                          (short)0, acc[1][j], false, false);
    }
  }

  if (valid) {
    const int rsub = (lane >> 4) << 3;
    const int csub = lane & 15;
    gf_p cg = (gf_p)(unsigned long long)C;
#pragma unroll
    for (int m2 = 0; m2 < 2; ++m2) {
      const int row0 = ((mt0 + m2) << 4) + rsub;
#pragma unroll
      for (int j = 0; j < 4; ++j) {
        gf_p cp = cg + (size_t)row0 * N + (size_t)((ntBase + j) << 4) + csub;
#pragma unroll
        for (int r = 0; r < 8; ++r) cp[(size_t)r * N] = acc[m2][j][r];
      }
    }
  }
}

// ---------------------------------------------------------------------------
// Depthwise causal conv (D_CONV=4) + bias + SiLU. Reads first half of xz.
// ---------------------------------------------------------------------------
__global__ void conv_silu_kernel(const float* __restrict__ xz, const float* __restrict__ cw,
                                 const float* __restrict__ cb, float* __restrict__ xb, int L) {
  int idx = blockIdx.x * blockDim.x + threadIdx.x;
  if (idx >= NBAT * L * DM) return;
  int d = idx & (DM - 1);
  int row = idx >> 9;
  int b = row / L, l = row % L;
  float acc = cb[d];
#pragma unroll
  for (int j = 0; j < 4; ++j) {
    int ll = l - 3 + j;
    if (ll >= 0) acc += cw[d * 4 + j] * xz[((size_t)(b * L + ll)) * 1024 + d];
  }
  xb[(size_t)row * DM + d] = silu_f(acc);
}

// ---------------------------------------------------------------------------
// Selective scan. One block = 16 states x 16 channels (256 threads).
// Reduction over n via 16-lane xor shuffles (wave32: masks <= 8 stay in-group).
// ---------------------------------------------------------------------------
__global__ void scan_kernel(const float* __restrict__ dtmm, const float* __restrict__ dtb,
                            const float* __restrict__ xdbl, const float* __restrict__ xb,
                            const float* __restrict__ Alog, float* __restrict__ yraw, int L) {
  int n  = threadIdx.x & 15;
  int dl = threadIdx.x >> 4;
  int dchunk = blockIdx.x & ((DM / 16) - 1);
  int b = blockIdx.x / (DM / 16);
  int d = dchunk * 16 + dl;
  float Av   = -expf(Alog[d * NST + n]);
  float bias = dtb[d];
  float h = 0.f;
  for (int l = 0; l < L; ++l) {
    size_t row = (size_t)b * L + l;
    if (l + 8 < L) {                          // hide latency behind exp/fma chain
      __builtin_prefetch(&dtmm[(row + 8) * DM + d], 0, 0);
      __builtin_prefetch(&xb[(row + 8) * DM + d], 0, 0);
      __builtin_prefetch(&xdbl[(row + 8) * 64 + DTR + n], 0, 0);
    }
    float dtv = softplus_f(dtmm[row * DM + d] + bias);
    float Bv  = xdbl[row * 64 + DTR + n];
    float Cv  = xdbl[row * 64 + DTR + NST + n];
    float xv  = xb[row * DM + d];
    h = expf(dtv * Av) * h + dtv * Bv * xv;
    float c = h * Cv;
    c += __shfl_xor(c, 1);
    c += __shfl_xor(c, 2);
    c += __shfl_xor(c, 4);
    c += __shfl_xor(c, 8);
    if (n == 0) yraw[row * DM + d] = c;
  }
}

// ---------------------------------------------------------------------------
// y2 = (yraw + xb*D_param) * silu(z), z = xz[:, 512+d]
// ---------------------------------------------------------------------------
__global__ void ymod_kernel(const float* __restrict__ yraw, const float* __restrict__ xb,
                            const float* __restrict__ Dp, const float* __restrict__ xz,
                            float* __restrict__ y2f, int M) {
  int idx = blockIdx.x * blockDim.x + threadIdx.x;
  if (idx >= M * DM) return;
  int d = idx & (DM - 1);
  int row = idx >> 9;
  float z = xz[(size_t)row * 1024 + DM + d];
  y2f[idx] = (yraw[idx] + xb[idx] * Dp[d]) * silu_f(z);
}

// ---------------------------------------------------------------------------
// x_full[b,d,l] = scale[d] * out[(b*L+l),d] * mask + xprev[b,d,l]
// ---------------------------------------------------------------------------
__global__ void residual_kernel(const float* __restrict__ outp, const float* __restrict__ xprev,
                                const float* __restrict__ scale, const unsigned char* __restrict__ mb,
                                int shift, float* __restrict__ xfull, int L) {
  int idx = blockIdx.x * blockDim.x + threadIdx.x;
  if (idx >= NBAT * DM * L) return;
  int l = idx % L;
  int d = (idx / L) & (DM - 1);
  int b = idx / (DM * L);
  float mval = mb[b * 2048 + (l << shift)] ? 1.f : 0.f;
  float v = scale[d] * outp[((size_t)(b * L + l)) * DM + d] * mval + xprev[idx];
  xfull[idx] = v;
}

// ---------------------------------------------------------------------------
// Max-pool window 3 stride 2 pad 1 over L, then mask[::2].
// ---------------------------------------------------------------------------
__global__ void pool_kernel(const float* __restrict__ xfull, const unsigned char* __restrict__ mb,
                            int shift, float* __restrict__ xnext, float* __restrict__ dfeat, int L) {
  int Lh = L >> 1;
  int idx = blockIdx.x * blockDim.x + threadIdx.x;
  if (idx >= NBAT * DM * Lh) return;
  int lp = idx % Lh;
  int d = (idx / Lh) & (DM - 1);
  int b = idx / (DM * Lh);
  const float* base = xfull + (size_t)b * DM * L + (size_t)d * L;
  float m = -INFINITY;
#pragma unroll
  for (int t = -1; t <= 1; ++t) {
    int ll = 2 * lp + t;
    if (ll >= 0 && ll < L) m = fmaxf(m, base[ll]);
  }
  float mval = mb[b * 2048 + ((2 * lp) << shift)] ? 1.f : 0.f;
  float v = m * mval;
  xnext[idx] = v;
  dfeat[idx] = v;
}

// ---------------------------------------------------------------------------
// Mask outputs (3 scales) as floats, concatenated.
// ---------------------------------------------------------------------------
__global__ void mask_out_kernel(const unsigned char* __restrict__ mb, float* __restrict__ out) {
  int idx = blockIdx.x * blockDim.x + threadIdx.x;
  if (idx >= 4096 + 2048 + 1024) return;
  int i, Li, r;
  if (idx < 4096)      { i = 0; Li = 2048; r = idx; }
  else if (idx < 6144) { i = 1; Li = 1024; r = idx - 4096; }
  else                 { i = 2; Li = 512;  r = idx - 6144; }
  int b = r / Li, l = r % Li;
  out[idx] = mb[b * 2048 + (l << i)] ? 1.f : 0.f;
}

// ---------------------------------------------------------------------------

static inline int cdiv(int a, int b) { return (a + b - 1) / b; }

extern "C" void kernel_launch(void* const* d_in, const int* in_sizes, int n_in,
                              void* d_out, int out_size, void* d_ws, size_t ws_size,
                              hipStream_t stream) {
  (void)in_sizes; (void)n_in; (void)out_size; (void)ws_size;
  const int L0 = 2048;
  const int M0 = NBAT * L0;   // 4096

  const float* x_in = (const float*)d_in[0];
  const unsigned char* mb = (const unsigned char*)d_in[1];

  // workspace allocator (256B aligned)
  char* wp = (char*)d_ws;
  auto alloc = [&](size_t bytes) -> void* {
    void* r = (void*)wp;
    wp += (bytes + 255) & ~(size_t)255;
    return r;
  };
  float*  hln   = (float*) alloc((size_t)M0 * DM * 4);
  bf16_t* Aln   = (bf16_t*)alloc((size_t)M0 * DM * 2);
  float*  xz    = (float*) alloc((size_t)M0 * 1024 * 4);
  float*  xbf   = (float*) alloc((size_t)M0 * DM * 4);
  bf16_t* xbp   = (bf16_t*)alloc((size_t)M0 * DM * 2);
  float*  xdbl  = (float*) alloc((size_t)M0 * 64 * 4);
  bf16_t* dtp   = (bf16_t*)alloc((size_t)M0 * DTR * 2);
  float*  dtmm  = (float*) alloc((size_t)M0 * DM * 4);
  float*  yraw  = (float*) alloc((size_t)M0 * DM * 4);
  float*  y2f   = (float*) alloc((size_t)M0 * DM * 4);
  bf16_t* y2p   = (bf16_t*)alloc((size_t)M0 * DM * 2);
  float*  outp  = (float*) alloc((size_t)M0 * DM * 4);
  float*  xfull = (float*) alloc((size_t)M0 * DM * 4);
  float*  xnext = (float*) alloc((size_t)(M0 / 2) * DM * 4);
  bf16_t* wpin  = (bf16_t*)alloc((size_t)1024 * DM * 2);
  bf16_t* wpxp  = (bf16_t*)alloc((size_t)64 * DM * 2);
  bf16_t* wpdt  = (bf16_t*)alloc((size_t)DM * DTR * 2);
  bf16_t* wpout = (bf16_t*)alloc((size_t)DM * DM * 2);

  // d_out layout: feat0(2*512*2048) feat1(/2) feat2(/4) m0(4096) m1(2048) m2(1024)
  float* dout = (float*)d_out;
  const size_t f0 = (size_t)NBAT * DM * L0;
  hipMemcpyAsync(dout, x_in, f0 * sizeof(float), hipMemcpyDeviceToDevice, stream);
  float* dfeat = dout + f0;

  const float* xcur = x_in;
  int L = L0;
  for (int blk = 0; blk < 2; ++blk) {
    const int pb = 2 + blk * 12;
    const float* ln_g   = (const float*)d_in[pb + 0];
    const float* ln_b   = (const float*)d_in[pb + 1];
    const float* w_in   = (const float*)d_in[pb + 2];   // (1024, 512)
    const float* conv_w = (const float*)d_in[pb + 3];   // (512, 4)
    const float* conv_b = (const float*)d_in[pb + 4];
    const float* w_xp   = (const float*)d_in[pb + 5];   // (64, 512)
    const float* w_dt   = (const float*)d_in[pb + 6];   // (512, 32)
    const float* dt_b   = (const float*)d_in[pb + 7];
    const float* A_log  = (const float*)d_in[pb + 8];   // (512, 16)
    const float* D_par  = (const float*)d_in[pb + 9];
    const float* w_out  = (const float*)d_in[pb + 10];  // (512, 512)
    const float* scale  = (const float*)d_in[pb + 11];

    const int M = NBAT * L;

    // weight packs (deterministic each call; tiny)
    pack_b_kernel<<<cdiv(1024 * DM, 256), 256, 0, stream>>>(w_in,  wpin,  1024, DM);
    pack_b_kernel<<<cdiv(64 * DM,   256), 256, 0, stream>>>(w_xp,  wpxp,  64,   DM);
    pack_b_kernel<<<cdiv(DM * DTR,  256), 256, 0, stream>>>(w_dt,  wpdt,  DM,   DTR);
    pack_b_kernel<<<cdiv(DM * DM,   256), 256, 0, stream>>>(w_out, wpout, DM,   DM);

    // 1) LayerNorm
    ln_kernel<<<cdiv(M, 128), 128, 0, stream>>>(xcur, ln_g, ln_b, hln, L);
    pack_a_kernel<<<cdiv(M * DM, 256), 256, 0, stream>>>(hln, Aln, M, DM, DM);

    // 2) in_proj GEMM: (M,512) x (512,1024)   [block tile 32x256]
    wmma_gemm_bf16<<<dim3(M / 32, cdiv(1024, 256)), 128, 0, stream>>>(Aln, wpin, xz, M, 1024, DM);

    // 3) conv + silu
    conv_silu_kernel<<<cdiv(M * DM, 256), 256, 0, stream>>>(xz, conv_w, conv_b, xbf, L);
    pack_a_kernel<<<cdiv(M * DM, 256), 256, 0, stream>>>(xbf, xbp, M, DM, DM);

    // 4) x_proj GEMM: (M,512) x (512,64)
    wmma_gemm_bf16<<<dim3(M / 32, 1), 128, 0, stream>>>(xbp, wpxp, xdbl, M, 64, DM);

    // 5) dt_proj GEMM: (M,32) x (32,512)
    pack_a_kernel<<<cdiv(M * DTR, 256), 256, 0, stream>>>(xdbl, dtp, M, DTR, 64);
    wmma_gemm_bf16<<<dim3(M / 32, cdiv(DM, 256)), 128, 0, stream>>>(dtp, wpdt, dtmm, M, DM, DTR);

    // 6) selective scan (sequential in L)
    scan_kernel<<<NBAT * (DM / 16), 256, 0, stream>>>(dtmm, dt_b, xdbl, xbf, A_log, yraw, L);

    // 7) gate + D skip
    ymod_kernel<<<cdiv(M * DM, 256), 256, 0, stream>>>(yraw, xbf, D_par, xz, y2f, M);
    pack_a_kernel<<<cdiv(M * DM, 256), 256, 0, stream>>>(y2f, y2p, M, DM, DM);

    // 8) out_proj GEMM: (M,512) x (512,512)
    wmma_gemm_bf16<<<dim3(M / 32, cdiv(DM, 256)), 128, 0, stream>>>(y2p, wpout, outp, M, DM, DM);

    // 9) scale + mask + residual, then pool (writes next x and d_out feature)
    residual_kernel<<<cdiv(NBAT * DM * L, 256), 256, 0, stream>>>(outp, xcur, scale, mb, blk, xfull, L);
    pool_kernel<<<cdiv(NBAT * DM * (L / 2), 256), 256, 0, stream>>>(xfull, mb, blk, xnext, dfeat, L);

    dfeat += (size_t)NBAT * DM * (L / 2);
    xcur = xnext;
    L >>= 1;
  }

  // masks
  float* dmask = dout + f0 + f0 / 2 + f0 / 4;
  mask_out_kernel<<<cdiv(4096 + 2048 + 1024, 256), 256, 0, stream>>>(mb, dmask);
}
